// Network_75960791597069
// MI455X (gfx1250) — matrix operand order repaired and checked by reading, not compiled, not run on hardware
//
#include <hip/hip_runtime.h>
#include <math.h>

typedef __attribute__((ext_vector_type(16))) _Float16 v16h;
typedef __attribute__((ext_vector_type(8)))  _Float16 v8h;
typedef __attribute__((ext_vector_type(8)))  float    v8f;

// Force DS (addrspace 3) lowering for LDS stores the inference pass misses.
#define LDS_PTR(T, p) ((__attribute__((address_space(3))) T*)(p))

#define WAVES_PER_BLOCK 2

#define C_SQRT3  1.7320508075688772f
#define C_ISQRT3 0.5773502691896258f
#define C_ISQRT2 0.7071067811865475f
#define C_PI     3.141592653589793f
#define C_MAXR   3.5f
#define C_INVNB  0.17677669529663687f    /* 1/sqrt(32)  */
#define C_ISQ32  0.17677669529663687f
#define C_CS     0.3826834323650898f     /* sin(pi/8)   */
#define C_CX     0.9238795325112867f     /* cos(pi/8)   */
#define C_EMBN   2.823462196724446f      /* sqrt(10)/1.12 */
#define C_IS10   0.31622776601683794f    /* 1/sqrt(10)  */
#define C_IS100  0.1f                    /* 1/sqrt(100) */
#define C_INODE  0.028284271247461901f   /* 1/sqrt(1250)*/

#define WAVE_SYNC() do { __builtin_amdgcn_sched_barrier(0); \
                         __builtin_amdgcn_wave_barrier();   \
                         __builtin_amdgcn_sched_barrier(0); } while (0)

__device__ __forceinline__ int amapK(int h, int hi) {
  // 16-bit A-matrix 16x32 VGPR layout (ISA 7.12.2)
  return (h < 8) ? (8 * hi + h) : (16 + 8 * hi + (h - 8));
}

__device__ __forceinline__ v8f wmma16(v16h a, v16h b, v8f c) {
  return __builtin_amdgcn_wmma_f32_16x16x32_f16(false, a, false, b, (short)0, c,
                                                false, false);
}

// fast-math activations: native v_exp_f32 / v_rcp_f32 / v_cos_f32 TRANS ops
// (co-execute with WMMA on CDNA5); precision is ample for gated-GNN gates.
__device__ __forceinline__ float rcpf_fast(float x) { return __builtin_amdgcn_rcpf(x); }
__device__ __forceinline__ float sigmf(float x) {
  return rcpf_fast(1.0f + __expf(-x));
}
__device__ __forceinline__ float siluf(float x) { return x * sigmf(x); }

// Per-wave scratch for one 16-edge tile.
struct EdgeSmem {
  float    emb[16][10];
  float    a0[16];
  float    a1[16][3];
  float    w[16][80];
  _Float16 hidden[16][128];
};

// Per-block cache of radial-MLP weights, pre-converted to f16 and stored in
// WMMA B-fragment lane order: fragment read = two aligned ds_load_b128.
struct __align__(32) WCache {
  _Float16 b1f[7][32][16];      // GEMM1: W1 (10x100), K padded to 32, 7 N-tiles
  _Float16 b2f[5][4][32][16];   // GEMM2: W2 (100xP),  4 K-steps, <=5 N-tiles
};

__device__ void build_wcache(WCache& wc,
                             const float* __restrict__ W1,
                             const float* __restrict__ W2,
                             int P, int pTiles)
{
  for (int idx = threadIdx.x; idx < 7 * 32 * 16; idx += blockDim.x) {
    const int h    = idx & 15;
    const int lane = (idx >> 4) & 31;
    const int t    = idx >> 9;
    const int col  = t * 16 + (lane & 15);
    const int K    = amapK(h, lane >> 4);
    const float wv = (K < 10 && col < 100) ? W1[K * 100 + col] * C_IS10 : 0.0f;
    *LDS_PTR(_Float16, &wc.b1f[t][lane][h]) = (_Float16)wv;
  }
  for (int idx = threadIdx.x; idx < pTiles * 4 * 32 * 16; idx += blockDim.x) {
    const int h    = idx & 15;
    const int lane = (idx >> 4) & 31;
    const int ks   = (idx >> 9) & 3;
    const int t2   = idx >> 11;
    const int col  = t2 * 16 + (lane & 15);
    const int K    = ks * 32 + amapK(h, lane >> 4);
    const float wv = (K < 100 && col < P) ? W2[K * P + col] * C_IS100 : 0.0f;
    *LDS_PTR(_Float16, &wc.b2f[t2][ks][lane][h]) = (_Float16)wv;
  }
}

// ---------------------------------------------------------------------------
// Edge geometry + radial MLP (WMMA):  emb(16x10) @ W1(10x100) -> silu ->
// @ W2(100xP).  Results (a0,a1,w) land in per-wave LDS.
// ---------------------------------------------------------------------------
__device__ void edge_mlp(EdgeSmem& sm, const WCache& wc, int lane, int tile,
                         const float* __restrict__ edge_vec,
                         int P, int pTiles)
{
  const int l16 = lane & 15;
  const int hi  = lane >> 4;

  if (hi == 0) {
    const int   e  = tile * 16 + l16;
    const float vx = edge_vec[3*e+0], vy = edge_vec[3*e+1], vz = edge_vec[3*e+2];
    const float r   = sqrtf(vx*vx + vy*vy + vz*vz);
    const float inv = rcpf_fast(r + 1e-12f);
    const float sx = C_SQRT3 * vx * inv;
    const float sy = C_SQRT3 * vy * inv;
    const float sz = C_SQRT3 * vz * inv;
    const float u  = 2.0f * (r / C_MAXR - 1.0f);
    float cut;
    if (u > 0.0f)        cut = 0.0f;
    else if (u < -1.0f)  cut = 1.0f;
    else                 cut = 0.5f * (1.0f - __cosf(C_PI * u));
    *LDS_PTR(float, &sm.a0[l16])    = cut;
    *LDS_PTR(float, &sm.a1[l16][0]) = cut * sx;
    *LDS_PTR(float, &sm.a1[l16][1]) = cut * sy;
    *LDS_PTR(float, &sm.a1[l16][2]) = cut * sz;
    const float step = C_MAXR / 9.0f;
    #pragma unroll
    for (int k = 0; k < 10; ++k) {
      const float d = (r - step * (float)k) / step;
      *LDS_PTR(float, &sm.emb[l16][k]) = __expf(-d * d) * C_EMBN;
    }
  }
  // zero the hidden pad columns 112..127 (cols 0..111 written by GEMM1)
  #pragma unroll
  for (int j = 0; j < 8; ++j)
    *LDS_PTR(_Float16, &sm.hidden[l16][112 + 8 * hi + j]) = (_Float16)0.0f;
  WAVE_SYNC();

  // ---- GEMM1: A = emb tile (K padded 10 -> 32) ----
  v16h afrag;
  #pragma unroll
  for (int h = 0; h < 16; ++h) {
    const int K = amapK(h, hi);
    afrag[h] = (K < 10) ? (_Float16)sm.emb[l16][K] : (_Float16)0.0f;
  }
  for (int t = 0; t < 7; ++t) {
    const int col = t * 16 + l16;
    const v16h bfrag = *(const v16h*)(&wc.b1f[t][lane][0]);
    v8f acc = {};
    acc = wmma16(afrag, bfrag, acc);
    #pragma unroll
    for (int rr = 0; rr < 8; ++rr)
      *LDS_PTR(_Float16, &sm.hidden[rr + 8 * hi][col]) = (_Float16)siluf(acc[rr]);
  }
  WAVE_SYNC();

  // ---- GEMM2: hidden(16x100, padded 128) @ W2(100xP) ----
  for (int t2 = 0; t2 < pTiles; ++t2) {
    const int col = t2 * 16 + l16;
    v8f acc = {};
    #pragma unroll
    for (int ks = 0; ks < 4; ++ks) {
      // A fragment: two contiguous 16B runs of the row-major hidden tile
      const v8h alo = *(const v8h*)(&sm.hidden[l16][ks * 32 + 8 * hi]);
      const v8h ahi = *(const v8h*)(&sm.hidden[l16][ks * 32 + 16 + 8 * hi]);
      v16h a2;
      #pragma unroll
      for (int h = 0; h < 8; ++h) { a2[h] = alo[h]; a2[h + 8] = ahi[h]; }
      const v16h b2 = *(const v16h*)(&wc.b2f[t2][ks][lane][0]);
      acc = wmma16(a2, b2, acc);
    }
    #pragma unroll
    for (int rr = 0; rr < 8; ++rr)
      *LDS_PTR(float, &sm.w[rr + 8 * hi][col]) = acc[rr];
  }
  WAVE_SYNC();
}

// ---------------------------------------------------------------------------
// Layer-1 edge conv:  messages m0 (0e) / m1 (1o), scatter to g0[n], g1[n*3]
// ---------------------------------------------------------------------------
__global__ void edge_layer1_kernel(const float* __restrict__ edge_vec,
                                   const int*  __restrict__ eidx, int E, int eTiles,
                                   const float* __restrict__ W1,
                                   const float* __restrict__ W2,
                                   const float* __restrict__ W_lin1_1,
                                   float* __restrict__ g0, float* __restrict__ g1)
{
  __shared__ EdgeSmem smem[WAVES_PER_BLOCK];
  __shared__ WCache wc;
  const int wave = threadIdx.x >> 5, lane = threadIdx.x & 31;
  const int tile = blockIdx.x * WAVES_PER_BLOCK + wave;
  __builtin_prefetch(edge_vec + (size_t)48 * tile, 0, 3);
  __builtin_prefetch(eidx + E + 16 * tile, 0, 3);
  build_wcache(wc, W1, W2, 2, 1);
  __syncthreads();
  if (tile >= eTiles) return;
  EdgeSmem& sm = smem[wave];
  edge_mlp(sm, wc, lane, tile, edge_vec, 2, 1);

  if (lane < 16) {
    const int el  = lane;
    const int e   = tile * 16 + el;
    const int dst = eidx[E + e];
    const float xe = W_lin1_1[0];            // x == ones -> xl is a constant
    const float w0 = sm.w[el][0], w1 = sm.w[el][1];
    atomicAdd(&g0[dst], w0 * xe * sm.a0[el] * C_INVNB);
    const float c = w1 * xe * C_INVNB;
    atomicAdd(&g1[3*dst+0], c * sm.a1[el][0]);
    atomicAdd(&g1[3*dst+1], c * sm.a1[el][1]);
    atomicAdd(&g1[3*dst+2], c * sm.a1[el][2]);
  }
}

// ---------------------------------------------------------------------------
// Layer-2 edge conv: 5 TP paths, scatter into g00/g01/g10/g110/g111
// ---------------------------------------------------------------------------
__global__ void edge_layer2_kernel(const float* __restrict__ edge_vec,
                                   const int*  __restrict__ eidx, int E, int eTiles,
                                   const float* __restrict__ W1,
                                   const float* __restrict__ W2,
                                   const float* __restrict__ b0,
                                   const float* __restrict__ b1,
                                   float* __restrict__ g00, float* __restrict__ g01,
                                   float* __restrict__ g10, float* __restrict__ g110,
                                   float* __restrict__ g111)
{
  __shared__ EdgeSmem smem[WAVES_PER_BLOCK];
  __shared__ WCache wc;
  const int wave = threadIdx.x >> 5, lane = threadIdx.x & 31;
  const int tile = blockIdx.x * WAVES_PER_BLOCK + wave;
  __builtin_prefetch(edge_vec + (size_t)48 * tile, 0, 3);
  __builtin_prefetch(eidx + 16 * tile, 0, 3);
  build_wcache(wc, W1, W2, 80, 5);
  __syncthreads();
  if (tile >= eTiles) return;
  EdgeSmem& sm = smem[wave];
  edge_mlp(sm, wc, lane, tile, edge_vec, 80, 5);

  const int u = lane & 15, hi = lane >> 4;
  for (int it = 0; it < 8; ++it) {
    const int el  = hi + 2 * it;
    const int e   = tile * 16 + el;
    const int src = eidx[e], dst = eidx[E + e];
    const float a0v = sm.a0[el];
    const float a1x = sm.a1[el][0], a1y = sm.a1[el][1], a1z = sm.a1[el][2];
    const float w1v = sm.w[el][u],      w2v = sm.w[el][16+u];
    const float w3v = sm.w[el][32+u],   w4v = sm.w[el][48+u];
    const float w5v = sm.w[el][64+u];
    const float e0  = b0[16*src + u];
    const float e1x = b1[48*src + 3*u + 0];
    const float e1y = b1[48*src + 3*u + 1];
    const float e1z = b1[48*src + 3*u + 2];

    atomicAdd(&g00[16*dst + u], w1v * e0 * a0v * C_INVNB);
    const float k01 = w2v * e0 * C_INVNB;
    atomicAdd(&g01[48*dst + 3*u + 0], k01 * a1x);
    atomicAdd(&g01[48*dst + 3*u + 1], k01 * a1y);
    atomicAdd(&g01[48*dst + 3*u + 2], k01 * a1z);
    const float k10 = w3v * a0v * C_INVNB;
    atomicAdd(&g10[48*dst + 3*u + 0], k10 * e1x);
    atomicAdd(&g10[48*dst + 3*u + 1], k10 * e1y);
    atomicAdd(&g10[48*dst + 3*u + 2], k10 * e1z);
    const float dot = e1x*a1x + e1y*a1y + e1z*a1z;
    atomicAdd(&g110[16*dst + u], w4v * dot * C_ISQRT3 * C_INVNB);
    const float cxv = e1y*a1z - e1z*a1y;
    const float cyv = e1z*a1x - e1x*a1z;
    const float czv = e1x*a1y - e1y*a1x;
    const float k111 = w5v * C_ISQRT2 * C_INVNB;
    atomicAdd(&g111[48*dst + 3*u + 0], k111 * cxv);
    atomicAdd(&g111[48*dst + 3*u + 1], k111 * cyv);
    atomicAdd(&g111[48*dst + 3*u + 2], k111 * czv);
  }
}

// ---------------------------------------------------------------------------
// Final edge conv: m00 + m110, scatter into gmid[n*32]
// ---------------------------------------------------------------------------
__global__ void edge_final_kernel(const float* __restrict__ edge_vec,
                                  const int*  __restrict__ eidx, int E, int eTiles,
                                  const float* __restrict__ W1,
                                  const float* __restrict__ W2,
                                  const float* __restrict__ c0,
                                  const float* __restrict__ c1,
                                  float* __restrict__ gmid)
{
  __shared__ EdgeSmem smem[WAVES_PER_BLOCK];
  __shared__ WCache wc;
  const int wave = threadIdx.x >> 5, lane = threadIdx.x & 31;
  const int tile = blockIdx.x * WAVES_PER_BLOCK + wave;
  __builtin_prefetch(edge_vec + (size_t)48 * tile, 0, 3);
  __builtin_prefetch(eidx + 16 * tile, 0, 3);
  build_wcache(wc, W1, W2, 32, 2);
  __syncthreads();
  if (tile >= eTiles) return;
  EdgeSmem& sm = smem[wave];
  edge_mlp(sm, wc, lane, tile, edge_vec, 32, 2);

  const int u = lane & 15, hi = lane >> 4;
  for (int it = 0; it < 8; ++it) {
    const int el  = hi + 2 * it;
    const int e   = tile * 16 + el;
    const int src = eidx[e], dst = eidx[E + e];
    const float a0v = sm.a0[el];
    const float a1x = sm.a1[el][0], a1y = sm.a1[el][1], a1z = sm.a1[el][2];
    const float w00 = sm.w[el][u], w110 = sm.w[el][16+u];
    const float c0v = c0[16*src + u];
    const float dot = c1[48*src + 3*u + 0]*a1x +
                      c1[48*src + 3*u + 1]*a1y +
                      c1[48*src + 3*u + 2]*a1z;
    atomicAdd(&gmid[32*dst + u],      w00  * c0v * a0v * C_INVNB);
    atomicAdd(&gmid[32*dst + 16 + u], w110 * dot * C_ISQRT3 * C_INVNB);
  }
}

// ---------------------------------------------------------------------------
// Generic node linear: out[n,P] (+)= in[n,K] @ W[K,P] * scale  (K<=32)
// Strided addressing handles scalar channels and vector components alike.
// One wave per (16-node tile, 16-col tile).
// ---------------------------------------------------------------------------
__global__ void node_gemm_kernel(const float* __restrict__ in, int ldIn, int csIn,
                                 const float* __restrict__ Wm, int ldW,
                                 float* __restrict__ out, int ldOut, int csOut,
                                 int K, int P, float scale, int acc,
                                 int nTiles, int pTiles)
{
  const int wid = blockIdx.x * (blockDim.x >> 5) + (threadIdx.x >> 5);
  if (wid >= nTiles * pTiles) return;
  const int tN = wid % nTiles;
  const int tP = wid / nTiles;
  const int lane = threadIdx.x & 31, l16 = lane & 15, hi = lane >> 4;

  v16h a, b;
  #pragma unroll
  for (int h = 0; h < 16; ++h) {
    const int Kv = amapK(h, hi);
    const float v = (Kv < K) ? in[(tN * 16 + l16) * ldIn + Kv * csIn] : 0.0f;
    a[h] = (_Float16)v;
  }
  const int col = tP * 16 + l16;
  #pragma unroll
  for (int h = 0; h < 16; ++h) {
    const int Kv = amapK(h, hi);
    const float v = (Kv < K && col < P) ? Wm[Kv * ldW + col] * scale : 0.0f;
    b[h] = (_Float16)v;
  }
  v8f c = {};
  if (acc && col < P) {
    #pragma unroll
    for (int rr = 0; rr < 8; ++rr)
      c[rr] = out[(tN * 16 + rr + 8 * hi) * ldOut + col * csOut];
  }
  v8f d = wmma16(a, b, c);
  if (col < P) {
    #pragma unroll
    for (int rr = 0; rr < 8; ++rr)
      out[(tN * 16 + rr + 8 * hi) * ldOut + col * csOut] = d[rr];
  }
}

// ---------------------------------------------------------------------------
// Layer-1 gate (s_s / y_s are rank-1, handled elementwise)
// ---------------------------------------------------------------------------
__global__ void node1_kernel(int n, const float* __restrict__ g0,
                             const float* __restrict__ g1,
                             const float* __restrict__ Wss, const float* __restrict__ Wsg,
                             const float* __restrict__ Wls, const float* __restrict__ Wlg,
                             const float* __restrict__ Wlv,
                             float* __restrict__ x0, float* __restrict__ v1o)
{
  const int id = blockIdx.x * blockDim.x + threadIdx.x;
  if (id >= n * 16) return;
  const int node = id >> 4, u = id & 15;
  const float g0v  = g0[node];
  const float scal = C_CS * Wss[u] + C_CX * (g0v * Wls[u]);
  const float gts  = C_CS * Wsg[u] + C_CX * (g0v * Wlg[u]);
  const float gate = sigmf(gts);
  x0[16*node + u] = siluf(scal);
  const float wv = Wlv[u];
  v1o[48*node + 3*u + 0] = gate * g1[3*node + 0] * wv;
  v1o[48*node + 3*u + 1] = gate * g1[3*node + 1] * wv;
  v1o[48*node + 3*u + 2] = gate * g1[3*node + 2] * wv;
}

// ---------------------------------------------------------------------------
// Layer-2 skip-connect + gate
// ---------------------------------------------------------------------------
__global__ void combine2_kernel(int n,
                                const float* __restrict__ ss2, const float* __restrict__ sg2,
                                const float* __restrict__ sv2, const float* __restrict__ ys2,
                                const float* __restrict__ yg2, const float* __restrict__ yvo,
                                const float* __restrict__ yve,
                                float* __restrict__ x02, float* __restrict__ v1o2,
                                float* __restrict__ v1e2)
{
  const int id = blockIdx.x * blockDim.x + threadIdx.x;
  if (id >= n * 32) return;
  const int node = id >> 5, u = id & 31;
  const float gate = sigmf(C_CS * sg2[32*node + u] + C_CX * yg2[32*node + u]);
  if (u < 16) {
    x02[16*node + u] = siluf(C_CS * ss2[16*node + u] + C_CX * ys2[16*node + u]);
    #pragma unroll
    for (int i = 0; i < 3; ++i) {
      const float vo = C_CS * sv2[48*node + 3*u + i] + C_CX * yvo[48*node + 3*u + i];
      v1o2[48*node + 3*u + i] = gate * vo;
    }
  } else {
    const int uu = u - 16;
    #pragma unroll
    for (int i = 0; i < 3; ++i)
      v1e2[48*node + 3*uu + i] = gate * yve[48*node + 3*uu + i];
  }
}

// ---------------------------------------------------------------------------
// Final node output + per-graph reduction
// ---------------------------------------------------------------------------
__global__ void final_node_kernel(int n, const float* __restrict__ x02,
                                  const float* __restrict__ gmid,
                                  const float* __restrict__ Wscf,
                                  const float* __restrict__ Wlof,
                                  const int* __restrict__ batch,
                                  float* __restrict__ out)
{
  const int id = blockIdx.x * blockDim.x + threadIdx.x;
  if (id >= n) return;
  float sf = 0.0f;
  #pragma unroll
  for (int u = 0; u < 16; ++u) sf += x02[16*id + u] * Wscf[u];
  sf *= 0.25f;                                    // 1/sqrt(16)
  float y = 0.0f;
  #pragma unroll
  for (int k = 0; k < 32; ++k) y += gmid[32*id + k] * Wlof[k];
  y *= C_ISQ32;                                   // 1/sqrt(32)
  atomicAdd(&out[batch[id]], (C_CS * sf + C_CX * y) * C_INODE);
}

// ---------------------------------------------------------------------------
// Host side
// ---------------------------------------------------------------------------
static inline void ng(const float* in, int ldIn, int csIn,
                      const float* Wm, int ldW,
                      float* out, int ldOut, int csOut,
                      int K, int P, float scale, int acc, int n, hipStream_t s)
{
  const int nT = n / 16;
  const int pT = (P + 15) / 16;
  const int total = nT * pT;
  node_gemm_kernel<<<dim3((total + 3) / 4), dim3(128), 0, s>>>(
      in, ldIn, csIn, Wm, ldW, out, ldOut, csOut, K, P, scale, acc, nT, pT);
}

extern "C" void kernel_launch(void* const* d_in, const int* in_sizes, int n_in,
                              void* d_out, int out_size, void* d_ws, size_t ws_size,
                              hipStream_t stream)
{
  (void)n_in; (void)ws_size;
  const float* edge_vec = (const float*)d_in[1];
  const int*   eidx     = (const int*)d_in[2];
  const int*   batch    = (const int*)d_in[3];
  const float* W_sc1_s  = (const float*)d_in[4];
  const float* W_sc1_g  = (const float*)d_in[5];
  const float* W_lin1_1 = (const float*)d_in[6];
  const float* fc1_W1   = (const float*)d_in[7];
  const float* fc1_W2   = (const float*)d_in[8];
  const float* W_lo1_s  = (const float*)d_in[9];
  const float* W_lo1_g  = (const float*)d_in[10];
  const float* W_lo1_v  = (const float*)d_in[11];
  const float* W_sc2_s  = (const float*)d_in[12];
  const float* W_sc2_g  = (const float*)d_in[13];
  const float* W_sc2_v  = (const float*)d_in[14];
  const float* W_l1_2a  = (const float*)d_in[15];
  const float* W_l1_2b  = (const float*)d_in[16];
  const float* fc2_W1   = (const float*)d_in[17];
  const float* fc2_W2   = (const float*)d_in[18];
  const float* W_lo2_s  = (const float*)d_in[19];
  const float* W_lo2_g  = (const float*)d_in[20];
  const float* W_lo2_vo = (const float*)d_in[21];
  const float* W_lo2_ve = (const float*)d_in[22];
  const float* W_scf    = (const float*)d_in[23];
  const float* W_l1_fa  = (const float*)d_in[24];
  const float* W_l1_fb  = (const float*)d_in[25];
  /* d_in[26] = W_l1_fc: computed-but-unused in the reference */
  const float* fcf_W1   = (const float*)d_in[27];
  const float* fcf_W2   = (const float*)d_in[28];
  const float* W_lof    = (const float*)d_in[29];

  const int n = in_sizes[3];          // nodes (20000)
  const int E = in_sizes[1] / 3;      // edges (640000)
  const size_t N = (size_t)n;
  float* ws = (float*)d_ws;

  float* g0   = ws +   0*N;  float* g1   = ws +   1*N;
  float* x01  = ws +   4*N;  float* v1o1 = ws +  20*N;
  float* b0   = ws +  68*N;  float* b1   = ws +  84*N;
  float* ss2  = ws + 132*N;  float* sg2  = ws + 148*N;  float* sv2  = ws + 180*N;
  float* g00  = ws + 228*N;  float* g01  = ws + 244*N;  float* g10  = ws + 292*N;
  float* g110 = ws + 340*N;  float* g111 = ws + 356*N;
  float* ys2  = ws + 404*N;  float* yg2  = ws + 420*N;
  float* yvo  = ws + 452*N;  float* yve  = ws + 500*N;
  float* x02  = ws + 548*N;  float* v1o2 = ws + 564*N;  float* v1e2 = ws + 612*N;
  float* c0   = ws + 660*N;  float* c1   = ws + 676*N;
  float* gmid = ws + 724*N;                      // total 756*N floats (~61 MB)

  hipMemsetAsync(g0,    0,   4*N*sizeof(float), stream);
  hipMemsetAsync(g00,   0, 176*N*sizeof(float), stream);
  hipMemsetAsync(gmid,  0,  32*N*sizeof(float), stream);
  hipMemsetAsync(d_out, 0, (size_t)out_size*sizeof(float), stream);

  const int  eT = E / 16;
  const dim3 eBlk(32 * WAVES_PER_BLOCK);
  const dim3 eGrd((eT + WAVES_PER_BLOCK - 1) / WAVES_PER_BLOCK);

  // ---- layer 1 ----
  edge_layer1_kernel<<<eGrd, eBlk, 0, stream>>>(edge_vec, eidx, E, eT,
                                                fc1_W1, fc1_W2, W_lin1_1, g0, g1);
  node1_kernel<<<(n*16 + 255)/256, 256, 0, stream>>>(n, g0, g1, W_sc1_s, W_sc1_g,
                                                     W_lo1_s, W_lo1_g, W_lo1_v,
                                                     x01, v1o1);

  // ---- layer 2 pre-linears ----
  ng(x01, 16, 1, W_l1_2a, 16, b0, 16, 1, 16, 16, 0.25f, 0, n, stream);
  for (int i = 0; i < 3; ++i)
    ng(v1o1 + i, 48, 3, W_l1_2b, 16, b1 + i, 48, 3, 16, 16, 0.25f, 0, n, stream);
  ng(x01, 16, 1, W_sc2_s, 16, ss2, 16, 1, 16, 16, 0.25f, 0, n, stream);
  ng(x01, 16, 1, W_sc2_g, 32, sg2, 32, 1, 16, 32, 0.25f, 0, n, stream);
  for (int i = 0; i < 3; ++i)
    ng(v1o1 + i, 48, 3, W_sc2_v, 16, sv2 + i, 48, 3, 16, 16, 0.25f, 0, n, stream);

  edge_layer2_kernel<<<eGrd, eBlk, 0, stream>>>(edge_vec, eidx, E, eT,
                                                fc2_W1, fc2_W2, b0, b1,
                                                g00, g01, g10, g110, g111);

  // ---- layer 2 post-linears (concat handled as accumulate pass) ----
  ng(g00,  16, 1, W_lo2_s,         16, ys2, 16, 1, 16, 16, C_ISQ32, 0, n, stream);
  ng(g110, 16, 1, W_lo2_s + 16*16, 16, ys2, 16, 1, 16, 16, C_ISQ32, 1, n, stream);
  ng(g00,  16, 1, W_lo2_g,         32, yg2, 32, 1, 16, 32, C_ISQ32, 0, n, stream);
  ng(g110, 16, 1, W_lo2_g + 16*32, 32, yg2, 32, 1, 16, 32, C_ISQ32, 1, n, stream);
  for (int i = 0; i < 3; ++i) {
    ng(g01  + i, 48, 3, W_lo2_vo,         16, yvo + i, 48, 3, 16, 16, C_ISQ32, 0, n, stream);
    ng(g10  + i, 48, 3, W_lo2_vo + 16*16, 16, yvo + i, 48, 3, 16, 16, C_ISQ32, 1, n, stream);
    ng(g111 + i, 48, 3, W_lo2_ve,         16, yve + i, 48, 3, 16, 16, 0.25f,   0, n, stream);
  }
  combine2_kernel<<<(n*32 + 255)/256, 256, 0, stream>>>(n, ss2, sg2, sv2, ys2,
                                                        yg2, yvo, yve,
                                                        x02, v1o2, v1e2);

  // ---- final conv ----
  ng(x02, 16, 1, W_l1_fa, 16, c0, 16, 1, 16, 16, 0.25f, 0, n, stream);
  for (int i = 0; i < 3; ++i)
    ng(v1o2 + i, 48, 3, W_l1_fb, 16, c1 + i, 48, 3, 16, 16, 0.25f, 0, n, stream);

  edge_final_kernel<<<eGrd, eBlk, 0, stream>>>(edge_vec, eidx, E, eT,
                                               fcf_W1, fcf_W2, c0, c1, gmid);

  final_node_kernel<<<(n + 255)/256, 256, 0, stream>>>(n, x02, gmid, W_scf, W_lof,
                                                       batch, (float*)d_out);
}
